// VMamba_Feature_Extractor_modified_57449482551775
// MI455X (gfx1250) — matrix operand order repaired and checked by reading, not compiled, not run on hardware
//
#include <hip/hip_runtime.h>
#include <hip/hip_bf16.h>
#include <math.h>

// ---------------------------------------------------------------------------
// VMamba feature extractor for MI455X (gfx1250, wave32).
//  - Linear2d GEMMs: v_wmma_f32_16x16x32_bf16, 16(M)x64(N) tile per wave,
//    operands pre-packed to zero-padded bf16 (W: [M,K32]; X: transposed
//    [N,K32]) so every fragment is a contiguous unconditional vector load.
//  - 4-direction selective scan: chunked scan (segment reduce + Kogge-Stone
//    LDS scan of (a,b) pairs), ys fused into the apply pass.
//  - dt/softplus/exp/B*xs fused; xs never materialized (index gathers).
// Workspace (floats), requires ws_size >= 192 MB:
//   o_x   [ 0M.. 4M)  residual stream
//   o_t0  [ 4M.. 8M)  temp
//   o_t1  [ 8M..12M)  temp
//   o_dbl [12M..16M)  x_dbl [B,4,R+2,L]
//   o_a   [16M..32M)  a -> ys (overwritten) -> MLP hidden (reused)
//   o_b   [32M..48M)  b ; during GEMMs (o_b dead): Xpt bf16 at +0, Wp at +14M
// ---------------------------------------------------------------------------

typedef __attribute__((ext_vector_type(16))) __bf16 bf16x16;
typedef __attribute__((ext_vector_type(8)))  __bf16 bf16x8;
typedef __attribute__((ext_vector_type(8)))  float  f32x8;

#define BT 8   // total batch (4 + 4)

__device__ __forceinline__ float gelu_f(float x) {
  float x3 = x * x * x;
  return 0.5f * x * (1.f + tanhf(0.7978845608028654f * (x + 0.044715f * x3)));
}
__device__ __forceinline__ float silu_f(float x) { return x / (1.f + __expf(-x)); }
__device__ __forceinline__ float softplus_f(float x) {
  return (x > 20.f) ? x : log1pf(__expf(x));
}
// cross-scan direction index transform: position l of xs[k] -> index into x[b,d,H*W]
__device__ __forceinline__ int xs_index(int k, int l, int H, int W, int L) {
  switch (k & 3) {
    case 0: return l;
    case 1: { int w = l / H, h = l - w * H; return h * W + w; }
    case 2: return L - 1 - l;
    default: { int lr = L - 1 - l; int w = lr / H, h = lr - w * H; return h * W + w; }
  }
}

// ---------------- elementwise / conv kernels ----------------

__global__ void concat_kernel(const float* __restrict__ x0, const float* __restrict__ x1,
                              float* __restrict__ out, int perHalf) {
  int idx = blockIdx.x * blockDim.x + threadIdx.x;
  if (idx >= 2 * perHalf) return;
  out[idx] = (idx < perHalf) ? x0[idx] : x1[idx - perHalf];
}

__global__ void conv3x3_kernel(const float* __restrict__ in, const float* __restrict__ w,
                               const float* __restrict__ bias, float* __restrict__ out,
                               int B, int Cin, int Cout, int Hin, int Win,
                               int stride, int Hout, int Wout) {
  int idx = blockIdx.x * blockDim.x + threadIdx.x;
  int total = B * Cout * Hout * Wout;
  if (idx >= total) return;
  int wo = idx % Wout; int t = idx / Wout;
  int ho = t % Hout;  t /= Hout;
  int co = t % Cout;  int b = t / Cout;
  float acc = bias ? bias[co] : 0.f;
  const float* wv = w + (size_t)co * Cin * 9;
  for (int ci = 0; ci < Cin; ++ci) {
    const float* ip = in + ((size_t)(b * Cin + ci)) * Hin * Win;
    #pragma unroll
    for (int kh = 0; kh < 3; ++kh) {
      int h = ho * stride - 1 + kh;
      if ((unsigned)h >= (unsigned)Hin) continue;
      #pragma unroll
      for (int kw = 0; kw < 3; ++kw) {
        int ww = wo * stride - 1 + kw;
        if ((unsigned)ww >= (unsigned)Win) continue;
        acc += ip[h * Win + ww] * wv[(ci * 3 + kh) * 3 + kw];
      }
    }
  }
  out[idx] = acc;
}

__global__ void dwconv_silu_kernel(const float* __restrict__ in, const float* __restrict__ w,
                                   float* __restrict__ out, int B, int C, int H, int W) {
  int idx = blockIdx.x * blockDim.x + threadIdx.x;
  int total = B * C * H * W;
  if (idx >= total) return;
  int wo = idx % W; int t = idx / W;
  int ho = t % H;  t /= H;
  int c = t % C;   int b = t / C;
  const float* ip = in + ((size_t)(b * C + c)) * H * W;
  const float* wv = w + c * 9;
  float acc = 0.f;
  #pragma unroll
  for (int kh = 0; kh < 3; ++kh) {
    int h = ho - 1 + kh;
    if ((unsigned)h >= (unsigned)H) continue;
    #pragma unroll
    for (int kw = 0; kw < 3; ++kw) {
      int ww = wo - 1 + kw;
      if ((unsigned)ww >= (unsigned)W) continue;
      acc += ip[h * W + ww] * wv[kh * 3 + kw];
    }
  }
  out[idx] = silu_f(acc);
}

__global__ void ln2d_kernel(const float* __restrict__ in, const float* __restrict__ g,
                            const float* __restrict__ be, float* __restrict__ out,
                            int B, int C, int HW, int act) {
  int idx = blockIdx.x * blockDim.x + threadIdx.x;
  if (idx >= B * HW) return;
  int hw = idx % HW; int b = idx / HW;
  const float* ip = in + (size_t)b * C * HW + hw;
  float m = 0.f;
  for (int c = 0; c < C; ++c) m += ip[(size_t)c * HW];
  m /= (float)C;
  float v = 0.f;
  for (int c = 0; c < C; ++c) { float d = ip[(size_t)c * HW] - m; v += d * d; }
  v /= (float)C;
  float r = rsqrtf(v + 1e-5f);
  float* op = out + (size_t)b * C * HW + hw;
  for (int c = 0; c < C; ++c) {
    float y = (ip[(size_t)c * HW] - m) * r * g[c] + be[c];
    if (act == 1) y = gelu_f(y);
    op[(size_t)c * HW] = y;
  }
}

// ---------------- GEMM operand packing ----------------

// W[M,K] f32 -> Wp[M,K32] bf16, zero-padded K
__global__ void pack_w_kernel(const float* __restrict__ W, __bf16* __restrict__ Wp,
                              int M, int K, int K32) {
  int idx = blockIdx.x * blockDim.x + threadIdx.x;
  if (idx >= M * K32) return;
  int kc = idx % K32; int m = idx / K32;
  Wp[idx] = (__bf16)((kc < K) ? W[(size_t)m * K + kc] : 0.f);
}

// X[b,K,N] f32 -> Xpt[b,N,K32] bf16 (transposed, zero-padded K); tiled via LDS
__global__ void pack_xt_kernel(const float* __restrict__ X, __bf16* __restrict__ Xpt,
                               int K, int K32, int N) {
  __shared__ __bf16 tile[32][33];
  int b = blockIdx.z;
  int n0 = blockIdx.x * 32, k0 = blockIdx.y * 32;
  const float* Xb = X + (size_t)b * K * N;
  #pragma unroll
  for (int i = threadIdx.y; i < 32; i += 8) {
    int k = k0 + i, n = n0 + threadIdx.x;
    float v = (k < K) ? Xb[(size_t)k * N + n] : 0.f;
    tile[i][threadIdx.x] = (__bf16)v;
  }
  __syncthreads();
  __bf16* Ob = Xpt + (size_t)b * N * K32;
  #pragma unroll
  for (int i = threadIdx.y; i < 32; i += 8) {
    int n = n0 + i, k = k0 + threadIdx.x;
    Ob[(size_t)n * K32 + k] = tile[threadIdx.x][i];
  }
}

// ---------------- WMMA GEMM ----------------
// out[b,m,n] = sum_k W[m,k]*X[b,k,n] (+bias)(gelu)(+add)
// One wave per 16(M)x64(N) tile; A fragment shared across 4 N-subtiles.
// A layout (16-bit A 16x32): lane half hl, per lane one contiguous
// K-run: [kk+16*hl .. +15] (matches B).
// B layout (32x16): lane half selects K 0-15 / 16-31 -> contiguous in Xpt rows.
__global__ void wmma_linear_kernel(const __bf16* __restrict__ Wp,
                                   const __bf16* __restrict__ Xpt,
                                   const float* __restrict__ bias,
                                   const float* __restrict__ add,
                                   float* __restrict__ out,
                                   int M, int K32, int N, int act) {
  int b = blockIdx.z;
  int m0 = blockIdx.y * 16;
  int n0 = blockIdx.x * 64;
  int lane = threadIdx.x;
  int hl = lane >> 4, l16 = lane & 15;
  const __bf16* Xb = Xpt + (size_t)b * N * K32;
  float* Ob = out + (size_t)b * M * N;
  int am = m0 + l16; if (am >= M) am = M - 1;     // clamp: invalid rows never stored
  const __bf16* wrow = Wp + (size_t)am * K32;
  const __bf16* xr = Xb + (size_t)(n0 + l16) * K32 + hl * 16;
  const size_t rs = (size_t)16 * K32;             // 16 N-rows stride
  f32x8 acc0 = {}, acc1 = {}, acc2 = {}, acc3 = {};
  for (int kk = 0; kk < K32; kk += 32) {
    bf16x8 alo = *(const bf16x8*)(wrow + kk + hl * 16);
    bf16x8 ahi = *(const bf16x8*)(wrow + kk + hl * 16 + 8);
    bf16x16 afrag = __builtin_shufflevector(alo, ahi, 0, 1, 2, 3, 4, 5, 6, 7,
                                            8, 9, 10, 11, 12, 13, 14, 15);
    bf16x16 b0 = *(const bf16x16*)(xr + kk);
    bf16x16 b1 = *(const bf16x16*)(xr + rs + kk);
    bf16x16 b2 = *(const bf16x16*)(xr + 2 * rs + kk);
    bf16x16 b3 = *(const bf16x16*)(xr + 3 * rs + kk);
    acc0 = __builtin_amdgcn_wmma_f32_16x16x32_bf16(false, afrag, false, b0, (short)0, acc0, false, false);
    acc1 = __builtin_amdgcn_wmma_f32_16x16x32_bf16(false, afrag, false, b1, (short)0, acc1, false, false);
    acc2 = __builtin_amdgcn_wmma_f32_16x16x32_bf16(false, afrag, false, b2, (short)0, acc2, false, false);
    acc3 = __builtin_amdgcn_wmma_f32_16x16x32_bf16(false, afrag, false, b3, (short)0, acc3, false, false);
  }
  #pragma unroll
  for (int r = 0; r < 8; ++r) {
    int m = m0 + hl * 8 + r;                      // C/D layout: vgpr r -> M
    if (m < M) {
      float bs = bias ? bias[m] : 0.f;
      size_t rowo = (size_t)m * N + n0 + l16;
      #pragma unroll
      for (int j = 0; j < 4; ++j) {
        float v = (j == 0 ? acc0[r] : j == 1 ? acc1[r] : j == 2 ? acc2[r] : acc3[r]) + bs;
        if (act == 1) v = gelu_f(v);
        size_t o = rowo + j * 16;
        if (add) v += add[(size_t)b * M * N + o];
        Ob[o] = v;
      }
    }
  }
}

// ---------------- SS2D pieces ----------------

__global__ void xproj_kernel(const float* __restrict__ x, const float* __restrict__ w,
                             float* __restrict__ xdbl, int Bt, int d, int H, int W, int C2) {
  int L = H * W;
  int idx = blockIdx.x * blockDim.x + threadIdx.x;
  if (idx >= Bt * 4 * L) return;
  int l = idx % L; int t = idx / L;
  int k = t % 4;   int b = t / 4;
  int gi = xs_index(k, l, H, W, L);
  const float* xb = x + (size_t)b * d * L;
  const float* wk = w + (size_t)k * C2 * d;
  float acc[8];
  #pragma unroll
  for (int c = 0; c < 8; ++c) acc[c] = 0.f;
  for (int dc = 0; dc < d; ++dc) {
    float xv = xb[(size_t)dc * L + gi];
    #pragma unroll
    for (int c = 0; c < 8; ++c) if (c < C2) acc[c] += wk[c * d + dc] * xv;
  }
  float* op = xdbl + ((size_t)(b * 4 + k) * C2) * L + l;
  #pragma unroll
  for (int c = 0; c < 8; ++c) if (c < C2) op[(size_t)c * L] = acc[c];
}

__global__ void ab_kernel(const float* __restrict__ x, const float* __restrict__ xdbl,
                          const float* __restrict__ dtw, const float* __restrict__ dtb,
                          const float* __restrict__ alogs,
                          float* __restrict__ av, float* __restrict__ bv,
                          int Bt, int d, int H, int W, int R) {
  int L = H * W, C2 = R + 2;
  int idx = blockIdx.x * blockDim.x + threadIdx.x;
  if (idx >= Bt * 4 * L) return;
  int l = idx % L; int t = idx / L;
  int k = t % 4;   int b = t / 4;
  const float* dblp = xdbl + ((size_t)(b * 4 + k) * C2) * L + l;
  float dts[6];
  #pragma unroll
  for (int r = 0; r < 6; ++r) if (r < R) dts[r] = dblp[(size_t)r * L];
  float Bs = dblp[(size_t)R * L];
  int gi = xs_index(k, l, H, W, L);
  const float* xb = x + (size_t)b * d * L;
  size_t rowbase = ((size_t)(b * 4 + k) * d) * L + l;
  for (int dc = 0; dc < d; ++dc) {
    float s = dtb[k * d + dc];
    const float* wv = dtw + ((size_t)(k * d + dc)) * R;
    #pragma unroll
    for (int r = 0; r < 6; ++r) if (r < R) s += wv[r] * dts[r];
    float dt = softplus_f(s);
    float A = -__expf(alogs[k * d + dc]);
    float xsv = xb[(size_t)dc * L + gi];
    av[rowbase + (size_t)dc * L] = __expf(dt * A);
    bv[rowbase + (size_t)dc * L] = dt * Bs * xsv;
  }
}

__global__ void scan_ys_kernel(float* __restrict__ av, const float* __restrict__ bv,
                               const float* __restrict__ xdbl, const float* __restrict__ x,
                               const float* __restrict__ Ds,
                               int Bt, int d, int H, int W, int R) {
  int L = H * W, C2 = R + 2;
  int row = blockIdx.x;               // row = (b*4 + k)*d + dc
  int t = threadIdx.x;                // 256 threads
  int dc = row % d; int k = (row / d) % 4; int b = row / (4 * d);
  size_t base = (size_t)row * L;
  int seg = L / 256;
  size_t s0 = base + (size_t)t * seg;
  float Ap = 1.f, Bp = 0.f;
  for (int i = 0; i < seg; ++i) { float a = av[s0 + i], bb = bv[s0 + i]; Ap *= a; Bp = a * Bp + bb; }
  __shared__ float sA[256], sB[256];
  sA[t] = Ap; sB[t] = Bp;
  __syncthreads();
  for (int off = 1; off < 256; off <<= 1) {
    float ca = sA[t], cb = sB[t], pa = 1.f, pb = 0.f;
    if (t >= off) { pa = sA[t - off]; pb = sB[t - off]; }
    __syncthreads();
    if (t >= off) { sA[t] = pa * ca; sB[t] = ca * pb + cb; }
    __syncthreads();
  }
  float h = (t == 0) ? 0.f : sB[t - 1];
  const float* Cp = xdbl + (((size_t)(b * 4 + k) * C2) + (R + 1)) * L;
  const float* xb = x + (size_t)b * d * L;
  float Dv = Ds[k * d + dc];
  for (int i = 0; i < seg; ++i) {
    int l = t * seg + i;
    float a = av[s0 + i], bb = bv[s0 + i];
    h = a * h + bb;
    int gi = xs_index(k, l, H, W, L);
    av[s0 + i] = Cp[l] * h + Dv * xb[(size_t)dc * L + gi];   // ys
  }
}

__global__ void merge_kernel(const float* __restrict__ ys, float* __restrict__ y,
                             int Bt, int d, int H, int W) {
  int L = H * W;
  int idx = blockIdx.x * blockDim.x + threadIdx.x;
  if (idx >= Bt * d * L) return;
  int hw = idx % L; int t = idx / L;
  int dc = t % d;   int b = t / d;
  int h = hw / W, w = hw - h * W;
  int lt = w * H + h;
  size_t s0 = ((size_t)(b * 4 + 0) * d + dc) * L;
  size_t s1 = ((size_t)(b * 4 + 1) * d + dc) * L;
  size_t s2 = ((size_t)(b * 4 + 2) * d + dc) * L;
  size_t s3 = ((size_t)(b * 4 + 3) * d + dc) * L;
  y[idx] = ys[s0 + hw] + ys[s2 + (L - 1 - hw)] + ys[s1 + lt] + ys[s3 + (L - 1 - lt)];
}

__global__ void split_copy_kernel(const float* __restrict__ feat, float* __restrict__ out,
                                  size_t off0, size_t off1, int C, int HW) {
  int per = C * HW;
  int idx = blockIdx.x * blockDim.x + threadIdx.x;
  if (idx >= 8 * per) return;
  int b = idx / per; int r = idx - b * per;
  if (b < 4) out[off0 + (size_t)b * per + r] = feat[idx];
  else       out[off1 + (size_t)(b - 4) * per + r] = feat[idx];
}

// ---------------- host orchestration ----------------

static inline int cdiv(int a, int b) { return (a + b - 1) / b; }

static void wmma_lin(const float* X, const float* W, const float* bias, const float* add,
                     float* out, int M, int K, int N, int act,
                     __bf16* wp, __bf16* xpt, hipStream_t s) {
  int K32 = (K + 31) & ~31;
  { int n = M * K32;
    pack_w_kernel<<<cdiv(n, 256), 256, 0, s>>>(W, wp, M, K, K32); }
  { dim3 g(N / 32, K32 / 32, BT); dim3 bl(32, 8);
    pack_xt_kernel<<<g, bl, 0, s>>>(X, xpt, K, K32, N); }
  { dim3 g(N / 64, cdiv(M, 16), BT);
    wmma_linear_kernel<<<g, 32, 0, s>>>(wp, xpt, bias, add, out, M, K32, N, act); }
}
static void run_ln(const float* in, const float* g, const float* be, float* out,
                   int C, int HW, int act, hipStream_t s) {
  int n = BT * HW;
  ln2d_kernel<<<cdiv(n, 256), 256, 0, s>>>(in, g, be, out, BT, C, HW, act);
}
static void run_conv(const float* in, const float* w, const float* b, float* out,
                     int Cin, int Cout, int Hin, int Win, int stride, hipStream_t s) {
  int Hout = Hin / stride, Wout = Win / stride;
  int n = BT * Cout * Hout * Wout;
  conv3x3_kernel<<<cdiv(n, 256), 256, 0, s>>>(in, w, b, out, BT, Cin, Cout, Hin, Win,
                                              stride, Hout, Wout);
}

extern "C" void kernel_launch(void* const* d_in, const int* in_sizes, int n_in,
                              void* d_out, int out_size, void* d_ws, size_t ws_size,
                              hipStream_t stream) {
  (void)in_sizes; (void)n_in; (void)out_size; (void)ws_size;
  float* ws = (float*)d_ws;
  const size_t M1 = 1u << 20;
  float* o_x   = ws;
  float* o_t0  = ws + 4 * M1;
  float* o_t1  = ws + 8 * M1;
  float* o_dbl = ws + 12 * M1;
  float* o_a   = ws + 16 * M1;
  float* o_b   = ws + 32 * M1;
  __bf16* o_xp = (__bf16*)o_b;                 // Xpt (o_b dead during GEMMs)
  __bf16* o_wp = (__bf16*)(o_b + 14 * M1);     // Wp
  float* outp  = (float*)d_out;

  int pi = 0;
  auto nxt = [&]() { return (const float*)d_in[pi++]; };
  const float* x0 = nxt();
  const float* x1 = nxt();
  const float *pe_w1 = nxt(), *pe_b1 = nxt(), *pe_g1 = nxt(), *pe_be1 = nxt();
  const float *pe_w2 = nxt(), *pe_b2 = nxt(), *pe_g2 = nxt(), *pe_be2 = nxt();

  // ---- patch embed ----
  {
    int n = BT * 256 * 256;
    concat_kernel<<<cdiv(n, 256), 256, 0, stream>>>(x0, x1, o_t0, 4 * 256 * 256);
  }
  run_conv(o_t0, pe_w1, pe_b1, o_x, 1, 12, 256, 256, 2, stream);
  run_ln(o_x, pe_g1, pe_be1, o_t0, 12, 128 * 128, 1, stream);        // ln + gelu
  run_conv(o_t0, pe_w2, pe_b2, o_t1, 12, 24, 128, 128, 1, stream);
  run_ln(o_t1, pe_g2, pe_be2, o_x, 24, 128 * 128, 0, stream);

  const int dims[3] = {24, 48, 96};
  const int deps[3] = {1, 2, 2};
  const size_t off0[3] = {0, 786432, 1179648};
  const size_t off1[3] = {1376256, 2162688, 2555904};

  int H = 128, W = 128;
  for (int s = 0; s < 3; ++s) {
    int d = dims[s];
    int R = (d + 15) / 16;
    int HW = H * W;
    for (int blk = 0; blk < deps[s]; ++blk) {
      const float *n1g = nxt(), *n1b = nxt(), *in_proj = nxt(), *conv_w = nxt();
      const float *x_proj_w = nxt(), *dt_w = nxt(), *dt_bias = nxt();
      const float *A_logs = nxt(), *Dsp = nxt(), *on_g = nxt(), *on_b = nxt();
      const float *out_proj = nxt(), *n2g = nxt(), *n2b = nxt();
      const float *fc1w = nxt(), *fc1b = nxt(), *fc2w = nxt(), *fc2b = nxt();

      // ---- SS2D ----
      run_ln(o_x, n1g, n1b, o_t0, d, HW, 0, stream);
      wmma_lin(o_t0, in_proj, nullptr, nullptr, o_t1, d, d, HW, 0, o_wp, o_xp, stream);
      {
        int n = BT * d * HW;
        dwconv_silu_kernel<<<cdiv(n, 256), 256, 0, stream>>>(o_t1, conv_w, o_t0, BT, d, H, W);
      }
      {
        int n = BT * 4 * HW;
        xproj_kernel<<<cdiv(n, 256), 256, 0, stream>>>(o_t0, x_proj_w, o_dbl,
                                                       BT, d, H, W, R + 2);
        ab_kernel<<<cdiv(n, 256), 256, 0, stream>>>(o_t0, o_dbl, dt_w, dt_bias, A_logs,
                                                    o_a, o_b, BT, d, H, W, R);
      }
      scan_ys_kernel<<<BT * 4 * d, 256, 0, stream>>>(o_a, o_b, o_dbl, o_t0, Dsp,
                                                     BT, d, H, W, R);
      {
        int n = BT * d * HW;
        merge_kernel<<<cdiv(n, 256), 256, 0, stream>>>(o_a, o_t1, BT, d, H, W);
      }
      run_ln(o_t1, on_g, on_b, o_t0, d, HW, 0, stream);
      wmma_lin(o_t0, out_proj, nullptr, o_x, o_x, d, d, HW, 0, o_wp, o_xp, stream);

      // ---- MLP ----
      run_ln(o_x, n2g, n2b, o_t0, d, HW, 0, stream);
      wmma_lin(o_t0, fc1w, fc1b, nullptr, o_a, 4 * d, d, HW, 1, o_wp, o_xp, stream);
      wmma_lin(o_a, fc2w, fc2b, o_x, o_x, d, 4 * d, HW, 0, o_wp, o_xp, stream);
    }
    // ---- downsample: conv3x3 s2 -> ln2d ----
    const float *dw = nxt(), *db = nxt(), *dg = nxt(), *dbe = nxt();
    run_conv(o_x, dw, db, o_t0, d, 2 * d, H, W, 2, stream);
    H /= 2; W /= 2;
    run_ln(o_t0, dg, dbe, o_x, 2 * d, H * W, 0, stream);
    {
      int n = BT * 2 * d * H * W;
      split_copy_kernel<<<cdiv(n, 256), 256, 0, stream>>>(o_x, outp, off0[s], off1[s],
                                                          2 * d, H * W);
    }
  }
}
